// Attention_19756849561949
// MI455X (gfx1250) — matrix operand order repaired and checked by reading, not compiled, not run on hardware
//
#include <hip/hip_runtime.h>

typedef __attribute__((ext_vector_type(16))) _Float16 v16h;
typedef __attribute__((ext_vector_type(8)))  _Float16 v8h;
typedef __attribute__((ext_vector_type(2)))  _Float16 h2;
typedef __attribute__((ext_vector_type(8)))  float    v8f;

static constexpr int kN     = 4096;  // sequence length
static constexpr int kD     = 64;    // head dim
static constexpr int kBK    = 64;    // keys per chunk
static constexpr int kWaves = 4;     // waves per block (256 blocks -> fills more WGPs)
static constexpr int kRowsW = 16;    // q rows per wave
static constexpr int kRowsB = kWaves * kRowsW; // 64 q rows per block

#if __has_builtin(__builtin_amdgcn_exp2f)
#define EXP2F(x) __builtin_amdgcn_exp2f(x)
#else
#define EXP2F(x) exp2f(x)
#endif

// Raw v_max without canonicalize glue around WMMA/asm values.
__device__ __forceinline__ float fmax_raw(float a, float b) {
    float r;
    asm("v_max_num_f32_e32 %0, %1, %2" : "=v"(r) : "v"(a), "v"(b));
    return r;
}

// Max-reduce across a 16-lane DPP row: 4 v_max ops with fused DPP on src0.
__device__ __forceinline__ float dppmax16(float x) {
    float r;
    asm("v_max_num_f32_dpp %0, %1, %1 quad_perm:[1,0,3,2] row_mask:0xf bank_mask:0xf bound_ctrl:1\n\t"
        "v_max_num_f32_dpp %0, %0, %0 quad_perm:[2,3,0,1] row_mask:0xf bank_mask:0xf bound_ctrl:1\n\t"
        "v_max_num_f32_dpp %0, %0, %0 row_half_mirror row_mask:0xf bank_mask:0xf bound_ctrl:1\n\t"
        "v_max_num_f32_dpp %0, %0, %0 row_mirror row_mask:0xf bank_mask:0xf bound_ctrl:1"
        : "=&v"(r) : "v"(x));
    return r;
}

// PV key permutation: slot s holds physical key chi(s) = (s&32) + ((s&1)<<4)
// + ((s>>1)&15). Lets the C->A P bounce pack two f16 per ds_store_b32 while
// the A-fragment load stays 2x contiguous b128. For slot bases that are a
// multiple of 32 this reduces to base + (j&1)*16 + (j>>1): compile-time
// offsets for the V staging loads.

__global__ __launch_bounds__(128)
void Attention_19756849561949_kernel(const float* __restrict__ q,
                                     const float* __restrict__ k,
                                     const float* __restrict__ v,
                                     float* __restrict__ out)
{
    // Double-buffered K/V staging (one barrier per iteration) + P bounce.
    __shared__ __align__(16) _Float16 sK [2][kBK * kD];           // 16 KB
    __shared__ __align__(16) _Float16 sVT[2][kD  * kBK];          // 16 KB (slot-permuted keys)
    __shared__ __align__(16) _Float16 sP [kWaves][kRowsW * kBK];  //  8 KB

    const int tid   = threadIdx.x;
    const int lane  = tid & 31;
    const int wave  = tid >> 5;
    const int half  = lane >> 4;
    const int l16   = lane & 15;
    const int b     = blockIdx.y;
    const int qBase = blockIdx.x * kRowsB + wave * kRowsW;

    const size_t batchOff = (size_t)b * kN * kD;
    const float* kbase = k + batchOff;
    const float* vbase = v + batchOff;

    // Q fragments in A layout; fold sqrt(N)=64 and log2(e) so the softmax
    // runs natively in the exp2 domain (bare v_exp_f32).
    const float qscale = 64.0f * 1.44269504088896340736f;
    v16h aQ[2];
    {
        const float* qrow = q + batchOff + (size_t)(qBase + l16) * kD;
        const int kb = half * 16;
        #pragma unroll
        for (int c = 0; c < 2; ++c) {
            #pragma unroll
            for (int i = 0; i < 8; ++i) {
                aQ[c][i]     = (_Float16)(qrow[c * 32 + kb + i]      * qscale);
                aQ[c][i + 8] = (_Float16)(qrow[c * 32 + kb + 8 + i] * qscale);
            }
        }
    }

    // Ones B-tile: wmma(P, ones) accumulates the softmax denominator with the
    // exact l = l*corr + rowsum recurrence, no lane reductions.
    v16h bOnes;
    #pragma unroll
    for (int i = 0; i < 16; ++i) bOnes[i] = (_Float16)1.0f;

    v8f acc[4], acc5;
    #pragma unroll
    for (int dt = 0; dt < 4; ++dt)
        #pragma unroll
        for (int r = 0; r < 8; ++r) acc[dt][r] = 0.0f;
    #pragma unroll
    for (int r = 0; r < 8; ++r) acc5[r] = 0.0f;

    float mrow[8];
    #pragma unroll
    for (int r = 0; r < 8; ++r) mrow[r] = -3.0e38f;

    // Per-thread staging slots: 32 f16 per buffer (4x ds_store_b128 each).
    const int skey = tid >> 1;           // K: key row (64 rows)
    const int sd0  = (tid & 1) * 32;     // K: 32 consecutive d
    const int svd  = tid >> 1;           // V^T: d row (64 rows)
    const int svk0 = (tid & 1) * 32;     // V^T: 32 consecutive slots

    // ---- prefetch chunk 0, converting to f16 immediately (32 VGPRs held) ----
    v8h hk[4], hv[4];
    {
        const float* src = kbase + (size_t)skey * kD + sd0;
        #pragma unroll
        for (int u = 0; u < 4; ++u) {
            const float4 f0 = *(const float4*)(src + 8 * u);
            const float4 f1 = *(const float4*)(src + 8 * u + 4);
            v8h h;
            h[0] = (_Float16)f0.x; h[1] = (_Float16)f0.y;
            h[2] = (_Float16)f0.z; h[3] = (_Float16)f0.w;
            h[4] = (_Float16)f1.x; h[5] = (_Float16)f1.y;
            h[6] = (_Float16)f1.z; h[7] = (_Float16)f1.w;
            hk[u] = h;
        }
        const float* vsrc = vbase + (size_t)svk0 * kD + svd;
        #pragma unroll
        for (int u = 0; u < 4; ++u) {
            v8h h;
            #pragma unroll
            for (int i = 0; i < 8; ++i) {
                const int j = u * 8 + i;
                h[i] = (_Float16)vsrc[((j & 1) * 16 + (j >> 1)) * kD]; // static offsets
            }
            hv[u] = h;
        }
    }

    int buf = 0;
    for (int kc = 0; kc < kN; kc += kBK, buf ^= 1) {
        // ---- store prefetched f16 -> LDS[buf] ----
        #pragma unroll
        for (int u = 0; u < 4; ++u) {
            *(v8h*)&sK [buf][skey * kD  + sd0  + 8 * u] = hk[u];
            *(v8h*)&sVT[buf][svd  * kBK + svk0 + 8 * u] = hv[u];
        }
        __syncthreads();

        // ---- prefetch next chunk (overlaps WMMA/softmax below) ----
        if (kc + kBK < kN) {
            const float* src = kbase + (size_t)(kc + kBK + skey) * kD + sd0;
            #pragma unroll
            for (int u = 0; u < 4; ++u) {
                const float4 f0 = *(const float4*)(src + 8 * u);
                const float4 f1 = *(const float4*)(src + 8 * u + 4);
                v8h h;
                h[0] = (_Float16)f0.x; h[1] = (_Float16)f0.y;
                h[2] = (_Float16)f0.z; h[3] = (_Float16)f0.w;
                h[4] = (_Float16)f1.x; h[5] = (_Float16)f1.y;
                h[6] = (_Float16)f1.z; h[7] = (_Float16)f1.w;
                hk[u] = h;
            }
            const float* vsrc = vbase + (size_t)(kc + kBK + svk0) * kD + svd;
            #pragma unroll
            for (int u = 0; u < 4; ++u) {
                v8h h;
                #pragma unroll
                for (int i = 0; i < 8; ++i) {
                    const int j = u * 8 + i;
                    h[i] = (_Float16)vsrc[((j & 1) * 16 + (j >> 1)) * kD];
                }
                hv[u] = h;
            }
        }

        // ---- S = (qscale*Q) K^T : four 16x16 tiles over 64 keys ----
        v8f s[4];
        #pragma unroll
        for (int t = 0; t < 4; ++t) {
            v8f c = {};
            #pragma unroll
            for (int cc = 0; cc < 2; ++cc) {
                v16h bK;
                const _Float16* src = &sK[buf][(t * 16 + l16) * kD + cc * 32 + half * 16];
                #pragma unroll
                for (int i = 0; i < 16; ++i) bK[i] = src[i];
                c = __builtin_amdgcn_wmma_f32_16x16x32_f16(
                        false, aQ[cc], false, bK, (short)0, c, false, false);
            }
            s[t] = c;
        }

        // ---- online softmax in exp2 domain ----
        float p[4][8];
        #pragma unroll
        for (int r = 0; r < 8; ++r) {
            const float m01 = fmax_raw(s[0][r], s[1][r]);
            const float m23 = fmax_raw(s[2][r], s[3][r]);
            const float mn  = fmax_raw(mrow[r], dppmax16(fmax_raw(m01, m23)));
            const float corr = EXP2F(mrow[r] - mn);
            mrow[r] = mn;
            #pragma unroll
            for (int t = 0; t < 4; ++t) p[t][r] = EXP2F(s[t][r] - mn);
            #pragma unroll
            for (int dt = 0; dt < 4; ++dt) acc[dt][r] *= corr;
            acc5[r] *= corr;
        }

        // ---- P: C-layout -> A-layout bounce, pair-packed b32 stores ----
        uint32_t* pw32 = (uint32_t*)&sP[wave][0];
        #pragma unroll
        for (int r = 0; r < 8; ++r) {
            const int m = r + half * 8;
            h2 a01 = { (_Float16)p[0][r], (_Float16)p[1][r] };
            h2 a23 = { (_Float16)p[2][r], (_Float16)p[3][r] };
            pw32[m * 32 + l16]      = __builtin_bit_cast(uint32_t, a01);
            pw32[m * 32 + 16 + l16] = __builtin_bit_cast(uint32_t, a23);
        }
        asm volatile("s_wait_dscnt 0" ::: "memory");

        const _Float16* pw = &sP[wave][0];
        v16h aP[2];
        #pragma unroll
        for (int c2 = 0; c2 < 2; ++c2) {
            const _Float16* src = &pw[l16 * kBK + c2 * 32 + half * 16];
            #pragma unroll
            for (int i = 0; i < 8; ++i) { aP[c2][i] = src[i]; aP[c2][i + 8] = src[8 + i]; }
        }

        // ---- O += P @ V ; denom += P @ ones ----
        #pragma unroll
        for (int dt = 0; dt < 4; ++dt) {
            #pragma unroll
            for (int c2 = 0; c2 < 2; ++c2) {
                v16h bV;
                const _Float16* src = &sVT[buf][(dt * 16 + l16) * kBK + c2 * 32 + half * 16];
                #pragma unroll
                for (int i = 0; i < 16; ++i) bV[i] = src[i];
                acc[dt] = __builtin_amdgcn_wmma_f32_16x16x32_f16(
                              false, aP[c2], false, bV, (short)0, acc[dt], false, false);
            }
        }
        acc5 = __builtin_amdgcn_wmma_f32_16x16x32_f16(
                   false, aP[0], false, bOnes, (short)0, acc5, false, false);
        acc5 = __builtin_amdgcn_wmma_f32_16x16x32_f16(
                   false, aP[1], false, bOnes, (short)0, acc5, false, false);
    }

    // ---- epilogue ----
    #pragma unroll
    for (int r = 0; r < 8; ++r) {
        const float inv = 1.0f / acc5[r];
        const int m = r + half * 8;
        float* orow = out + batchOff + (size_t)(qBase + m) * kD;
        #pragma unroll
        for (int dt = 0; dt < 4; ++dt)
            orow[dt * 16 + l16] = acc[dt][r] * inv;
    }
}

extern "C" void kernel_launch(void* const* d_in, const int* in_sizes, int n_in,
                              void* d_out, int out_size, void* d_ws, size_t ws_size,
                              hipStream_t stream) {
    const float* q = (const float*)d_in[0];
    const float* k = (const float*)d_in[1];
    const float* v = (const float*)d_in[2];
    float* out = (float*)d_out;
    const int B = in_sizes[0] / (kN * kD);
    dim3 grid(kN / kRowsB, B, 1);
    Attention_19756849561949_kernel<<<grid, dim3(128, 1, 1), 0, stream>>>(q, k, v, out);
}
